// MessagePaiNN_9689446220428
// MI455X (gfx1250) — matrix operand order repaired and checked by reading, not compiled, not run on hardware
//
#include <hip/hip_runtime.h>
#include <hip/hip_bf16.h>
#include <math.h>

typedef __attribute__((ext_vector_type(16))) _Float16 v16h;
typedef __attribute__((ext_vector_type(8)))  _Float16 v8h;
typedef __attribute__((ext_vector_type(8)))  float    v8f;
typedef __attribute__((ext_vector_type(4)))  float    v4f;

constexpr int   FDIM   = 128;
constexpr int   FDIM3  = 384;
constexpr int   NRBF_C = 20;
constexpr float CUTF   = 5.0f;
constexpr float PI_F   = 3.14159265358979f;
constexpr int   WPB    = 4;    // waves per block
constexpr int   EPW    = 16;   // edges per wave (= WMMA M)

// Packed-weight panel table (each panel: 32 lanes x 16 f16 = 1 KB)
constexpr int PANEL_ELEMS = 32 * 16;                    // 512 f16
constexpr int W1_PBASE = 0;                             // 8 tiles * 4 ksteps = 32
constexpr int W2_PBASE = 32;                            // 24 tiles * 4 ksteps = 96
constexpr int WR_PBASE = 128;                           // 24 tiles * 1 kstep  = 24
constexpr int NPANELS  = 152;
constexpr size_t PACKED_BYTES = (size_t)NPANELS * PANEL_ELEMS * sizeof(_Float16);

__device__ __forceinline__ v8f wmma_f16(v16h a, v16h b, v8f c) {
  return __builtin_amdgcn_wmma_f32_16x16x32_f16(false, a, false, b, (short)0, c, false, false);
}

__device__ __forceinline__ void atomic_add_f32(float* p, float v) {
  __hip_atomic_fetch_add(p, v, __ATOMIC_RELAXED, __HIP_MEMORY_SCOPE_AGENT);
}

// B operand (32x16): B[k][n] = W[k*ld + col0 + n].
// ISA layout: h=lane/16; element j<8 -> K=k0+8h+j, element j>=8 -> K=k0+16+8h+(j-8).
__device__ __forceinline__ v16h load_b_panel_direct(const float* __restrict__ W, int ld,
                                                    int col0, int k0, int lane, int kmax) {
  int n  = lane & 15;
  int kb = k0 + ((lane & 16) ? 8 : 0);
  v16h b;
#pragma unroll
  for (int j = 0; j < 16; ++j) {
    int k = kb + (j < 8 ? j : j + 8);
    float v = (k < kmax) ? W[(size_t)k * ld + col0 + n] : 0.0f;
    b[j] = (_Float16)v;
  }
  return b;
}

template <bool PACKED>
__device__ __forceinline__ v16h get_b(const float* __restrict__ W, int ld, int col0, int k0,
                                      int lane, int kmax,
                                      const _Float16* __restrict__ pw, int panel) {
  if (PACKED) {
    return *(const v16h*)(pw + (size_t)panel * PANEL_ELEMS + lane * 16);
  } else {
    return load_b_panel_direct(W, ld, col0, k0, lane, kmax);
  }
}

// A operand (16x32) from a contiguous f32 row (vectorized: 2x float4 per half).
// lane m=lane&15; lanes<16: elems 0-7 = K=k0..k0+7, elems 8-15 = K=k0+16..+23;
// lanes>=16: same with +8 K offset.
__device__ __forceinline__ v16h load_a_row_f32(const float* __restrict__ row, int k0, int lane) {
  int koff = (lane & 16) ? 8 : 0;
  v4f lo0 = *(const v4f*)(row + k0 + koff);
  v4f lo1 = *(const v4f*)(row + k0 + koff + 4);
  v4f hi0 = *(const v4f*)(row + k0 + 16 + koff);
  v4f hi1 = *(const v4f*)(row + k0 + 16 + koff + 4);
  v16h a;
#pragma unroll
  for (int j = 0; j < 4; ++j) {
    a[j]      = (_Float16)lo0[j];
    a[j + 4]  = (_Float16)lo1[j];
    a[j + 8]  = (_Float16)hi0[j];
    a[j + 12] = (_Float16)hi1[j];
  }
  return a;
}

__device__ __forceinline__ v16h load_a_row_f16(const _Float16* __restrict__ row, int k0, int lane) {
  int koff = (lane & 16) ? 8 : 0;
  v8h lo = *(const v8h*)(row + k0 + koff);
  v8h hi = *(const v8h*)(row + k0 + 16 + koff);
  v16h a;
#pragma unroll
  for (int j = 0; j < 8; ++j) {
    a[j]     = lo[j];
    a[j + 8] = hi[j];
  }
  return a;
}

// ---- Weight prepack: f32 weights -> f16 panels in exact WMMA B lane layout ----
__global__ void painn_pack_kernel(const float* __restrict__ W1,
                                  const float* __restrict__ W2,
                                  const float* __restrict__ Wr,
                                  _Float16* __restrict__ pw) {
  int tid = blockIdx.x * blockDim.x + threadIdx.x;   // one thread per (panel, lane)
  if (tid >= NPANELS * 32) return;
  int panel = tid >> 5;
  int lane  = tid & 31;
  const float* W; int ld, col0, k0, kmax;
  if (panel < W2_PBASE) {
    W = W1; ld = FDIM;  col0 = (panel >> 2) * 16; k0 = (panel & 3) * 32; kmax = FDIM;
  } else if (panel < WR_PBASE) {
    int q = panel - W2_PBASE;
    W = W2; ld = FDIM3; col0 = (q >> 2) * 16;     k0 = (q & 3) * 32;     kmax = FDIM;
  } else {
    int q = panel - WR_PBASE;
    W = Wr; ld = FDIM3; col0 = q * 16;            k0 = 0;                kmax = NRBF_C;
  }
  v16h b = load_b_panel_direct(W, ld, col0, k0, lane, kmax);
  *(v16h*)(pw + (size_t)panel * PANEL_ELEMS + lane * 16) = b;
}

// out = [node_scalar (N*F), node_vector (N*F*3)], vectorized float4 copy
__global__ void painn_init_kernel(const float* __restrict__ ns, const float* __restrict__ nv,
                                  float* __restrict__ out, int NF, int total4) {
  int i = blockIdx.x * blockDim.x + threadIdx.x;   // float4 index
  if (i >= total4) return;
  int base = i * 4;
  v4f v;
  if (base + 3 < NF) {
    v = *(const v4f*)(ns + base);
  } else if (base >= NF) {
    v = *(const v4f*)(nv + (base - NF));
  } else {
#pragma unroll
    for (int k = 0; k < 4; ++k) {
      int idx = base + k;
      v[k] = (idx < NF) ? ns[idx] : nv[idx - NF];
    }
  }
  *(v4f*)(out + base) = v;
}

template <bool PACKED>
__global__ __launch_bounds__(WPB * 32)
void painn_edge_kernel(const float* __restrict__ node_scalar,
                       const float* __restrict__ node_vector,
                       const float* __restrict__ r_vec,
                       const float* __restrict__ W1, const float* __restrict__ bias1,
                       const float* __restrict__ W2, const float* __restrict__ bias2,
                       const float* __restrict__ Wr, const float* __restrict__ biasr,
                       const int*   __restrict__ edge_idx,
                       const _Float16* __restrict__ pw,
                       float* __restrict__ out_s,   // N*F
                       float* __restrict__ out_v,   // N*F*3
                       int E) {
  __shared__ _Float16 hbuf[WPB * EPW * FDIM];   // post-SiLU hidden, f16 (16 KB)
  __shared__ int   s_src[WPB * EPW];
  __shared__ int   s_dst[WPB * EPW];
  __shared__ float s_d[WPB * EPW];
  __shared__ float s_fcut[WPB * EPW];
  __shared__ float s_rhat[WPB * EPW][3];

  const int lane  = threadIdx.x & 31;
  const int wave  = threadIdx.x >> 5;
  const int wbase = wave * EPW;
  const int ebase = (blockIdx.x * WPB + wave) * EPW;

  // ---- Phase 0: per-edge metadata (lanes 0..15 = one edge each) ----
  if (lane < EPW) {
    int  e     = ebase + lane;
    bool valid = (e < E);
    int  ec    = valid ? e : 0;
    int  dsti  = edge_idx[2 * ec + 0];
    int  srci  = edge_idx[2 * ec + 1];
    float rx = r_vec[3 * ec + 0];
    float ry = r_vec[3 * ec + 1];
    float rz = r_vec[3 * ec + 2];
    float d  = sqrtf(rx * rx + ry * ry + rz * rz);
    float fc = (d < CUTF) ? 0.5f * (__cosf(PI_F * d / CUTF) + 1.0f) : 0.0f;
    if (!valid) fc = 0.0f;          // zero contribution for padded edges
    float invd = 1.0f / d;
    s_src[wbase + lane]  = srci;
    s_dst[wbase + lane]  = dsti;
    s_d[wbase + lane]    = d;
    s_fcut[wbase + lane] = fc;
    s_rhat[wbase + lane][0] = rx * invd;
    s_rhat[wbase + lane][1] = ry * invd;
    s_rhat[wbase + lane][2] = rz * invd;
  }
  __syncthreads();

  const int n  = lane & 15;
  const int mo = (lane & 16) ? 8 : 0;

  // ---- GEMM1: h = silu(X @ W1 + b1), X = node_scalar[src] (16x128) ----
  const float* xrow = node_scalar + (size_t)s_src[wbase + (lane & 15)] * FDIM;
  v16h ax[4];
#pragma unroll
  for (int s = 0; s < 4; ++s) ax[s] = load_a_row_f32(xrow, s * 32, lane);

  for (int t = 0; t < 8; ++t) {
    v8f acc = {};
#pragma unroll
    for (int s = 0; s < 4; ++s) {
      v16h b = get_b<PACKED>(W1, FDIM, t * 16, s * 32, lane, FDIM,
                             pw, W1_PBASE + t * 4 + s);
      acc = wmma_f16(ax[s], b, acc);
    }
#pragma unroll
    for (int j = 0; j < 8; ++j) {
      int col = t * 16 + n;
      float x = acc[j] + bias1[col];
      float hv = __fdividef(x, 1.0f + __expf(-x));        // fast SiLU
      hbuf[(size_t)(wbase + mo + j) * FDIM + col] = (_Float16)hv;
    }
  }
  __syncthreads();

  // ---- A operands for GEMM2: h in f16 from LDS, sinc built in registers ----
  const _Float16* hrow = &hbuf[(size_t)(wbase + (lane & 15)) * FDIM];
  v16h ha[4];
#pragma unroll
  for (int s = 0; s < 4; ++s) ha[s] = load_a_row_f16(hrow, s * 32, lane);

  v16h sa;  // sinc(d, n=1..20) padded to K=32
  {
    float dm   = s_d[wbase + (lane & 15)];
    float invd = 1.0f / dm;
    int koff = (lane & 16) ? 8 : 0;
#pragma unroll
    for (int j = 0; j < 8; ++j) {
      int k0 = koff + j;
      int k1 = 16 + koff + j;
      float v0 = (k0 < NRBF_C) ? __sinf(dm * (float)(k0 + 1) * (PI_F / CUTF)) * invd : 0.0f;
      float v1 = (k1 < NRBF_C) ? __sinf(dm * (float)(k1 + 1) * (PI_F / CUTF)) * invd : 0.0f;
      sa[j]     = (_Float16)v0;
      sa[j + 8] = (_Float16)v1;
    }
  }

  // ---- Hoist per-edge epilogue metadata into registers (reused by 24 tiles) ----
  float fcut_r[8];
  int   dst_r[8];
  int   src_r[8];
  float rh_r[8][3];
#pragma unroll
  for (int j = 0; j < 8; ++j) {
    int m = mo + j;
    fcut_r[j]  = s_fcut[wbase + m];
    dst_r[j]   = s_dst[wbase + m];
    src_r[j]   = s_src[wbase + m];
    rh_r[j][0] = s_rhat[wbase + m][0];
    rh_r[j][1] = s_rhat[wbase + m][1];
    rh_r[j][2] = s_rhat[wbase + m][2];
  }

  // ---- GEMM2 middle third (s2): scatter-add to scalar output ----
  for (int t = 0; t < 8; ++t) {
    v8f acc = {};
#pragma unroll
    for (int s = 0; s < 4; ++s) {
      v16h b = get_b<PACKED>(W2, FDIM3, 128 + t * 16, s * 32, lane, FDIM,
                             pw, W2_PBASE + (8 + t) * 4 + s);
      acc = wmma_f16(ha[s], b, acc);
    }
    v8f racc = {};
    {
      v16h wb = get_b<PACKED>(Wr, FDIM3, 128 + t * 16, 0, lane, NRBF_C,
                              pw, WR_PBASE + 8 + t);
      racc = wmma_f16(sa, wb, racc);
    }
    int col = 128 + t * 16 + n;
    float bb2 = bias2[col];
    float bbr = biasr[col];
#pragma unroll
    for (int j = 0; j < 8; ++j) {
      float pre = (acc[j] + bb2) * (racc[j] + bbr) * fcut_r[j];
      atomic_add_f32(&out_s[(size_t)dst_r[j] * FDIM + (t * 16 + n)], pre);
    }
  }

  // ---- GEMM2 outer thirds (s1 & s3) pairwise: vector message ----
  for (int t = 0; t < 8; ++t) {
    v8f acc1 = {}, acc3 = {};
#pragma unroll
    for (int s = 0; s < 4; ++s) {
      v16h bp1 = get_b<PACKED>(W2, FDIM3, t * 16, s * 32, lane, FDIM,
                               pw, W2_PBASE + t * 4 + s);
      acc1 = wmma_f16(ha[s], bp1, acc1);
      v16h bp3 = get_b<PACKED>(W2, FDIM3, 256 + t * 16, s * 32, lane, FDIM,
                               pw, W2_PBASE + (16 + t) * 4 + s);
      acc3 = wmma_f16(ha[s], bp3, acc3);
    }
    v8f r1 = {}, r3 = {};
    {
      v16h w1b = get_b<PACKED>(Wr, FDIM3, t * 16, 0, lane, NRBF_C,
                               pw, WR_PBASE + t);
      r1 = wmma_f16(sa, w1b, r1);
      v16h w3b = get_b<PACKED>(Wr, FDIM3, 256 + t * 16, 0, lane, NRBF_C,
                               pw, WR_PBASE + 16 + t);
      r3 = wmma_f16(sa, w3b, r3);
    }
    int f = t * 16 + n;
    float b2a = bias2[f],       bra = biasr[f];
    float b2c = bias2[256 + f], brc = biasr[256 + f];
#pragma unroll
    for (int j = 0; j < 8; ++j) {
      float fc  = fcut_r[j];
      float s1v = (acc1[j] + b2a) * (r1[j] + bra) * fc;
      float s3v = (acc3[j] + b2c) * (r3[j] + brc) * fc;
      const float* nv = node_vector + ((size_t)src_r[j] * FDIM + f) * 3;
      float*       ov = out_v       + ((size_t)dst_r[j] * FDIM + f) * 3;
      atomic_add_f32(&ov[0], nv[0] * s1v + s3v * rh_r[j][0]);
      atomic_add_f32(&ov[1], nv[1] * s1v + s3v * rh_r[j][1]);
      atomic_add_f32(&ov[2], nv[2] * s1v + s3v * rh_r[j][2]);
    }
  }
}

extern "C" void kernel_launch(void* const* d_in, const int* in_sizes, int n_in,
                              void* d_out, int out_size, void* d_ws, size_t ws_size,
                              hipStream_t stream) {
  const float* node_scalar = (const float*)d_in[0];
  const float* node_vector = (const float*)d_in[1];
  const float* r_vec       = (const float*)d_in[2];
  const float* W1          = (const float*)d_in[3];
  const float* b1          = (const float*)d_in[4];
  const float* W2          = (const float*)d_in[5];
  const float* b2          = (const float*)d_in[6];
  const float* Wr          = (const float*)d_in[7];
  const float* br          = (const float*)d_in[8];
  const int*   edge_idx    = (const int*)d_in[9];
  float* out = (float*)d_out;

  const int NF = in_sizes[0];           // N * 128
  const int E  = in_sizes[2] / 3;       // edges
  const int total = NF * 4;             // N*F + N*F*3
  const int total4 = total / 4;         // NF is a multiple of 4 (F = 128)

  // out = concat(node_scalar, node_vector)
  painn_init_kernel<<<(total4 + 255) / 256, 256, 0, stream>>>(
      node_scalar, node_vector, out, NF, total4);

  const int edges_per_block = WPB * EPW;   // 64
  const int blocks = (E + edges_per_block - 1) / edges_per_block;

  if (ws_size >= PACKED_BYTES) {
    _Float16* pw = (_Float16*)d_ws;
    painn_pack_kernel<<<(NPANELS * 32 + 255) / 256, 256, 0, stream>>>(W1, W2, Wr, pw);
    painn_edge_kernel<true><<<blocks, WPB * 32, 0, stream>>>(
        node_scalar, node_vector, r_vec, W1, b1, W2, b2, Wr, br, edge_idx,
        pw, out, out + (size_t)NF, E);
  } else {
    painn_edge_kernel<false><<<blocks, WPB * 32, 0, stream>>>(
        node_scalar, node_vector, r_vec, W1, b1, W2, b2, Wr, br, edge_idx,
        nullptr, out, out + (size_t)NF, E);
  }
}
